// SpatialAttention_1125281432069
// MI455X (gfx1250) — compile-verified, matrix operand order checked by reading, not run on hardware
//
#include <hip/hip_runtime.h>
#include <hip/hip_bf16.h>
#include <math.h>

typedef _Float16 f16;
typedef __attribute__((ext_vector_type(16))) _Float16 v16h;
typedef __attribute__((ext_vector_type(8)))  _Float16 v8h;
typedef __attribute__((ext_vector_type(8)))  float    v8f;

#define D_MODEL 2048
#define SEQ     8320
#define BATCH   2
#define ROWS    (BATCH*SEQ)     // 16640
#define NHEADS  16
#define HDIM    128
#define FFRAMES 8
#define NFR     (BATCH*FFRAMES) // 16
#define TPF     (SEQ/FFRAMES)   // 1040
#define NLAT    16
#define PATCH   (TPF-NLAT)      // 1024
#define TPAD    1056            // 66*16, padded token dim
#define FHN     (NFR*NHEADS)    // 256

#define KPANEL  512             // K-panel width (halves) staged in LDS
#define ASTRIDE 520             // padded row stride: rows land 4 banks apart

__device__ __forceinline__ v8f vzero8() {
  v8f z;
#pragma unroll
  for (int i = 0; i < 8; ++i) z[i] = 0.f;
  return z;
}

// Load one 16-bit WMMA operand fragment (A or B role) for this lane.
// p points at (row_for_this_lane)*stride + k_block_base. Per ISA layout:
// lanes 0-15 hold K {0..7,16..23}; lanes 16-31 hold K {8..15,24..31}.
__device__ __forceinline__ v16h load_frag(const f16* __restrict__ p, int hi) {
  const int o = hi ? 8 : 0;
  v8h lo = *(const v8h*)(p + o);
  v8h hh = *(const v8h*)(p + o + 16);
  v16h r;
#pragma unroll
  for (int i = 0; i < 8; ++i) { r[i] = lo[i]; r[i + 8] = hh[i]; }
  return r;
}

__device__ __forceinline__ v8f wmma_f16(v16h a, v16h b, v8f c) {
  return __builtin_amdgcn_wmma_f32_16x16x32_f16(false, a, false, b, (short)0, c, false, false);
}

// Async global->LDS copy of one 16-byte chunk (CDNA5 ASYNCcnt path).
// ldsoff is the per-lane LDS byte offset (HW adds the wave's LDS_BASE).
__device__ __forceinline__ void async_g2l_b128(unsigned ldsoff, const void* g) {
  asm volatile("global_load_async_to_lds_b128 %0, %1, off"
               :: "v"(ldsoff), "v"((unsigned long long)g) : "memory");
}
__device__ __forceinline__ void wait_async0() {
  asm volatile("s_wait_asynccnt 0x0" ::: "memory");
}

// ---------------------------------------------------------------- convert
__global__ void cvt_f32_f16(const float* __restrict__ s, f16* __restrict__ d, size_t n) {
  size_t i = (size_t)blockIdx.x * 256 + threadIdx.x;
  if (i < n) d[i] = (f16)s[i];
}

// Zero the token-pad region (tokens 1040..1055) of K and V^T.
__global__ void zero_pads(f16* __restrict__ kh, f16* __restrict__ vth) {
  int i = blockIdx.x * 256 + threadIdx.x;
  const int per = FHN * NLAT * HDIM;
  if (i < per) {
    int fh  = i / (NLAT * HDIM);
    int rem = i % (NLAT * HDIM);
    int t   = TPF + rem / HDIM;
    int dd  = rem % HDIM;
    kh[((size_t)fh * TPAD + t) * HDIM + dd]  = (f16)0.f;
    vth[((size_t)fh * HDIM + dd) * TPAD + t] = (f16)0.f;
  }
}

// Stage one 16-row x KPANEL A-panel into LDS with async B128 copies.
// Each of the 256 threads moves 4 chunks of 8 halves.
__device__ __forceinline__ void stage_panel(f16 (*atile)[ASTRIDE],
                                            const f16* __restrict__ src_m0,
                                            int panel_k) {
  const int tid = threadIdx.x;
  const int r   = tid >> 4;           // 0..15
  const int c0  = tid & 15;
  const f16* src = src_m0 + (size_t)r * D_MODEL + panel_k;
  const unsigned ldsrow = (unsigned)(unsigned long long)(void*)&atile[r][0];
#pragma unroll
  for (int j = 0; j < 4; ++j) {
    const int chunk = c0 + j * 16;    // 0..63, 8 halves each
    async_g2l_b128(ldsrow + chunk * 16, src + chunk * 8);
  }
  wait_async0();
  __syncthreads();
}

// ------------------------------------------------- fused QKV projection
// Block: 16 rows x 2048 cols. sel = blockIdx.y (0=q,1=k,2=v).
__global__ void __launch_bounds__(256)
qkv_proj(const f16* __restrict__ xh,
         const f16* __restrict__ wqh, const f16* __restrict__ wkh, const f16* __restrict__ wvh,
         const float* __restrict__ bq, const float* __restrict__ bk, const float* __restrict__ bv,
         const float* __restrict__ gq, const float* __restrict__ gk,
         const float* __restrict__ freqs,
         f16* __restrict__ qh, f16* __restrict__ kh, f16* __restrict__ vth)
{
  const int sel  = blockIdx.y;
  const f16*   w    = sel == 0 ? wqh : (sel == 1 ? wkh : wvh);
  const float* bias = sel == 0 ? bq  : (sel == 1 ? bk  : bv);
  const int m0   = blockIdx.x * 16;
  const int lane = threadIdx.x & 31;
  const int wid  = threadIdx.x >> 5;
  const int l15  = lane & 15;
  const int hi   = (lane >> 4) & 1;
  const int cbase = wid * 256;

  __shared__ f16 atile[16][ASTRIDE];
  __shared__ float rowsum[16];

  v8f acc[16];
#pragma unroll
  for (int t = 0; t < 16; ++t) acc[t] = vzero8();

  const f16* src_m0 = xh + (size_t)m0 * D_MODEL;
  for (int p = 0; p < D_MODEL; p += KPANEL) {
    stage_panel(atile, src_m0, p);
    for (int kk = 0; kk < KPANEL; kk += 32) {
      v16h a = load_frag(&atile[l15][kk], hi);
#pragma unroll
      for (int t = 0; t < 16; ++t) {
        const int col = cbase + t * 16 + l15;
        v16h b = load_frag(w + (size_t)col * D_MODEL + p + kk, hi);
        acc[t] = wmma_f16(a, b, acc[t]);
      }
    }
    __syncthreads();                  // panel consumed before refill
  }

#pragma unroll
  for (int t = 0; t < 16; ++t) {
    const float bb = bias[cbase + t * 16 + l15];
#pragma unroll
    for (int r = 0; r < 8; ++r) acc[t][r] += bb;
  }

  if (sel < 2) {                       // RMSNorm over the full 2048-wide row
    if (threadIdx.x < 16) rowsum[threadIdx.x] = 0.f;
    __syncthreads();
#pragma unroll
    for (int r = 0; r < 8; ++r) {
      float p = 0.f;
#pragma unroll
      for (int t = 0; t < 16; ++t) p += acc[t][r] * acc[t][r];
      atomicAdd(&rowsum[r + 8 * hi], p);
    }
    __syncthreads();
    float scale_r[8];
#pragma unroll
    for (int r = 0; r < 8; ++r)
      scale_r[r] = rsqrtf(rowsum[r + 8 * hi] * (1.f / (float)D_MODEL) + 1e-6f);
    const float* g = sel == 0 ? gq : gk;
#pragma unroll
    for (int t = 0; t < 16; ++t) {
      const float gg = g[cbase + t * 16 + l15];
#pragma unroll
      for (int r = 0; r < 8; ++r) acc[t][r] *= scale_r[r] * gg;
    }
  }

  int tok_r[8], fr_r[8], srow[8];
#pragma unroll
  for (int r = 0; r < 8; ++r) {
    int m   = m0 + r + 8 * hi;
    int b   = m / SEQ;
    int s   = m % SEQ;
    srow[r] = s;
    tok_r[r] = s % TPF;
    fr_r[r]  = b * FFRAMES + s / TPF;
  }

  if (sel < 2) {                       // RoPE + store row-major [fh][tok][d]
    f16* dst = sel == 0 ? qh : kh;
    const int par = lane & 1;
#pragma unroll
    for (int t = 0; t < 16; ++t) {
      const int col = cbase + t * 16 + l15;
      const int fi  = (col & 127) >> 1;
#pragma unroll
      for (int r = 0; r < 8; ++r) {
        float self  = acc[t][r];
        float other = __shfl_xor(self, 1, 32);
        float fv = freqs[(size_t)srow[r] * (HDIM / 2) + fi];
        float sn, cs;
        __sincosf(fv, &sn, &cs);
        float res = (par == 0) ? (self * cs - other * sn)
                               : (other * sn + self * cs);
        int fh = fr_r[r] * NHEADS + (col >> 7);
        dst[((size_t)fh * TPAD + tok_r[r]) * HDIM + (col & 127)] = (f16)res;
      }
    }
  } else {                             // V: store transposed [fh][d][tok]
#pragma unroll
    for (int t = 0; t < 16; ++t) {
      const int col = cbase + t * 16 + l15;
#pragma unroll
      for (int r = 0; r < 8; ++r) {
        int fh = fr_r[r] * NHEADS + (col >> 7);
        vth[((size_t)fh * HDIM + (col & 127)) * TPAD + tok_r[r]] = (f16)acc[t][r];
      }
    }
  }
}

// --------------------------------------------------- flash attention core
// One wave handles 16 queries; processes keys in pairs of 16-token tiles.
__device__ __forceinline__ void
flash_wave(const f16* __restrict__ qh, const f16* __restrict__ kh,
           const f16* __restrict__ vth, f16* __restrict__ oh,
           int fh, int qbase, int n_keypairs, int mask_last,
           f16 (*sp)[32], int lane)
{
  const int l15 = lane & 15;
  const int hi  = (lane >> 4) & 1;

  const f16* qrow = qh + ((size_t)fh * TPAD + qbase + l15) * HDIM;
  v16h qa[4];
#pragma unroll
  for (int ks = 0; ks < 4; ++ks) qa[ks] = load_frag(qrow + ks * 32, hi);

  v8f o[8];
#pragma unroll
  for (int nt = 0; nt < 8; ++nt) o[nt] = vzero8();
  float mrow[8], lrow[8];
#pragma unroll
  for (int r = 0; r < 8; ++r) { mrow[r] = -3.0e38f; lrow[r] = 0.f; }

  const f16* kbase = kh  + (size_t)fh * TPAD * HDIM;
  const f16* vbase = vth + (size_t)fh * HDIM * TPAD;
  const float sc = 0.08838834764831845f;   // 1/sqrt(128)

  for (int kp = 0; kp < n_keypairs; ++kp) {
    const int t0 = kp * 32;
    const int last = mask_last && (kp == n_keypairs - 1);
    v8f s0 = vzero8(), s1 = vzero8();
    const f16* k0 = kbase + (size_t)(t0 + l15) * HDIM;
#pragma unroll
    for (int ks = 0; ks < 4; ++ks)
      s0 = wmma_f16(qa[ks], load_frag(k0 + ks * 32, hi), s0);
    if (!last) {
      const f16* k1 = kbase + (size_t)(t0 + 16 + l15) * HDIM;
#pragma unroll
      for (int ks = 0; ks < 4; ++ks)
        s1 = wmma_f16(qa[ks], load_frag(k1 + ks * 32, hi), s1);
    }

    float alpha[8];
#pragma unroll
    for (int r = 0; r < 8; ++r) {
      float a0 = s0[r] * sc;
      float a1 = last ? -3.0e38f : s1[r] * sc;
      float rm = fmaxf(a0, a1);
#pragma unroll
      for (int msk = 1; msk < 16; msk <<= 1)
        rm = fmaxf(rm, __shfl_xor(rm, msk, 32));
      float mnew = fmaxf(mrow[r], rm);
      float al   = __expf(mrow[r] - mnew);
      float p0   = __expf(a0 - mnew);
      float p1   = last ? 0.f : __expf(a1 - mnew);
      float ps   = p0 + p1;
#pragma unroll
      for (int msk = 1; msk < 16; msk <<= 1)
        ps += __shfl_xor(ps, msk, 32);
      lrow[r] = lrow[r] * al + ps;
      mrow[r] = mnew;
      alpha[r] = al;
      sp[r + 8 * hi][l15]      = (f16)p0;   // stage P tile (row-major) in LDS
      sp[r + 8 * hi][16 + l15] = (f16)p1;
    }
#pragma unroll
    for (int nt = 0; nt < 8; ++nt)
#pragma unroll
      for (int r = 0; r < 8; ++r) o[nt][r] *= alpha[r];

    // reload P as an A-fragment (wave-private LDS, in-order per wave)
    v16h pa = load_frag(&sp[l15][0], hi);
#pragma unroll
    for (int nt = 0; nt < 8; ++nt) {
      const f16* vp = vbase + (size_t)(nt * 16 + l15) * TPAD + t0;
      o[nt] = wmma_f16(pa, load_frag(vp, hi), o[nt]);
    }
  }

  const int fr = fh >> 4, h = fh & 15;
  const int b = fr >> 3, frl = fr & 7;
  float inv_l[8];
#pragma unroll
  for (int r = 0; r < 8; ++r) inv_l[r] = 1.f / lrow[r];
#pragma unroll
  for (int nt = 0; nt < 8; ++nt) {
    const int col = h * HDIM + nt * 16 + l15;
#pragma unroll
    for (int r = 0; r < 8; ++r) {
      int tokq = qbase + r + 8 * hi;
      size_t m = (size_t)b * SEQ + (size_t)frl * TPF + tokq;
      oh[m * D_MODEL + col] = (f16)(o[nt][r] * inv_l[r]);
    }
  }
}

__global__ void __launch_bounds__(256)
attn_patch(const f16* __restrict__ qh, const f16* __restrict__ kh,
           const f16* __restrict__ vth, f16* __restrict__ oh)
{
  __shared__ f16 sp[8][16][32];
  const int wid = threadIdx.x >> 5;
  const int qt  = blockIdx.y * 8 + wid;     // 0..63
  flash_wave(qh, kh, vth, oh, blockIdx.x, qt * 16,
             PATCH / 32, 0, sp[wid], threadIdx.x & 31);
}

__global__ void __launch_bounds__(32)
attn_latent(const f16* __restrict__ qh, const f16* __restrict__ kh,
            const f16* __restrict__ vth, f16* __restrict__ oh)
{
  __shared__ f16 sp[16][32];
  // 16 latent queries vs all 1040 keys: 32 full pairs + 1 masked pair (to 1056)
  flash_wave(qh, kh, vth, oh, blockIdx.x, PATCH,
             33, 1, sp, threadIdx.x & 31);
}

// ------------------------------------------------------ output projection
__global__ void __launch_bounds__(256)
out_proj(const f16* __restrict__ oh, const f16* __restrict__ woh,
         const float* __restrict__ bo, float* __restrict__ out)
{
  const int m0   = blockIdx.x * 16;
  const int lane = threadIdx.x & 31;
  const int wid  = threadIdx.x >> 5;
  const int l15  = lane & 15;
  const int hi   = (lane >> 4) & 1;
  const int cbase = wid * 256;

  __shared__ f16 atile[16][ASTRIDE];

  v8f acc[16];
#pragma unroll
  for (int t = 0; t < 16; ++t) acc[t] = vzero8();

  const f16* src_m0 = oh + (size_t)m0 * D_MODEL;
  for (int p = 0; p < D_MODEL; p += KPANEL) {
    stage_panel(atile, src_m0, p);
    for (int kk = 0; kk < KPANEL; kk += 32) {
      v16h a = load_frag(&atile[l15][kk], hi);
#pragma unroll
      for (int t = 0; t < 16; ++t) {
        const int col = cbase + t * 16 + l15;
        v16h b = load_frag(woh + (size_t)col * D_MODEL + p + kk, hi);
        acc[t] = wmma_f16(a, b, acc[t]);
      }
    }
    __syncthreads();
  }

#pragma unroll
  for (int t = 0; t < 16; ++t) {
    const int col = cbase + t * 16 + l15;
    const float bb = bo[col];
#pragma unroll
    for (int r = 0; r < 8; ++r) {
      size_t m = (size_t)(m0 + r + 8 * hi);
      out[m * D_MODEL + col] = acc[t][r] + bb;
    }
  }
}

// ------------------------------------------------------------------ host
extern "C" void kernel_launch(void* const* d_in, const int* in_sizes, int n_in,
                              void* d_out, int out_size, void* d_ws, size_t ws_size,
                              hipStream_t stream) {
  (void)in_sizes; (void)n_in; (void)out_size; (void)ws_size;
  const float* x     = (const float*)d_in[0];
  const float* wq    = (const float*)d_in[1];
  const float* bq    = (const float*)d_in[2];
  const float* wk    = (const float*)d_in[3];
  const float* bk    = (const float*)d_in[4];
  const float* wv    = (const float*)d_in[5];
  const float* bv    = (const float*)d_in[6];
  const float* wo    = (const float*)d_in[7];
  const float* bo    = (const float*)d_in[8];
  const float* gq    = (const float*)d_in[9];
  const float* gk    = (const float*)d_in[10];
  const float* freqs = (const float*)d_in[11];
  float* out = (float*)d_out;

  const size_t XH = (size_t)ROWS * D_MODEL;
  const size_t WH = (size_t)D_MODEL * D_MODEL;
  const size_t AH = (size_t)FHN * TPAD * HDIM;

  f16* p   = (f16*)d_ws;
  f16* xh  = p; p += XH;
  f16* wqh = p; p += WH;
  f16* wkh = p; p += WH;
  f16* wvh = p; p += WH;
  f16* woh = p; p += WH;
  f16* qh  = p; p += AH;
  f16* kh  = p; p += AH;
  f16* vth = p; p += AH;
  f16* oh  = p; p += XH;

  auto cvt = [&](const float* s, f16* d, size_t n) {
    cvt_f32_f16<<<dim3((unsigned)((n + 255) / 256)), 256, 0, stream>>>(s, d, n);
  };
  cvt(x,  xh,  XH);
  cvt(wq, wqh, WH);
  cvt(wk, wkh, WH);
  cvt(wv, wvh, WH);
  cvt(wo, woh, WH);

  zero_pads<<<dim3((FHN * NLAT * HDIM + 255) / 256), 256, 0, stream>>>(kh, vth);

  qkv_proj<<<dim3(ROWS / 16, 3), 256, 0, stream>>>(
      xh, wqh, wkh, wvh, bq, bk, bv, gq, gk, freqs, qh, kh, vth);

  attn_patch<<<dim3(FHN, 8), 256, 0, stream>>>(qh, kh, vth, oh);
  attn_latent<<<dim3(FHN), 32, 0, stream>>>(qh, kh, vth, oh);

  out_proj<<<dim3(ROWS / 16), 256, 0, stream>>>(oh, woh, bo, out);
}
